// AttentionDGCNN_8469675508117
// MI455X (gfx1250) — compile-verified, matrix operand order checked by reading, not run on hardware
//
#include <hip/hip_runtime.h>

// AttentionDGCNN forward for gfx1250 (MI455X).
// All dense contractions run through v_wmma_f32_16x16x32_f16 with fused
// epilogues. GEMM uses a 2x2 register-blocked wave tile (32x32/wave,
// 64x64/block) so each K-step does 4 WMMAs on 4 fragment loads.
// Workspace requirement: ~60 MB (layout in kernel_launch).

typedef __attribute__((ext_vector_type(16))) _Float16 v16h;
typedef __attribute__((ext_vector_type(8)))  _Float16 v8h;
typedef __attribute__((ext_vector_type(4)))  _Float16 v4h;
typedef __attribute__((ext_vector_type(8)))  float    v8f;

#define EP_NONE     0
#define EP_BIAS     1
#define EP_BN_LRELU 2
#define EP_BIAS_RES 3

#define BN_RSQ 0.99999499f   // 1/sqrt(1+1e-5)
#define NEG_SLOPE 0.2f
#define KNN 20

// Build a 16-f16 WMMA fragment from fp32 memory: lane's data is two
// contiguous 8-float runs at p and p+16 (p must be 16B aligned).
__device__ __forceinline__ v16h load_frag_f32(const float* __restrict__ p)
{
    const float4 a = ((const float4*)p)[0];
    const float4 b = ((const float4*)p)[1];
    const float4 c = ((const float4*)(p + 16))[0];
    const float4 d = ((const float4*)(p + 16))[1];
    v16h v;
    v[0]  = (_Float16)a.x; v[1]  = (_Float16)a.y; v[2]  = (_Float16)a.z; v[3]  = (_Float16)a.w;
    v[4]  = (_Float16)b.x; v[5]  = (_Float16)b.y; v[6]  = (_Float16)b.z; v[7]  = (_Float16)b.w;
    v[8]  = (_Float16)c.x; v[9]  = (_Float16)c.y; v[10] = (_Float16)c.z; v[11] = (_Float16)c.w;
    v[12] = (_Float16)d.x; v[13] = (_Float16)d.y; v[14] = (_Float16)d.z; v[15] = (_Float16)d.w;
    return v;
}

// Same fragment shape from f16 memory (LDS): two 16B runs at p and p+16.
__device__ __forceinline__ v16h load_frag_h16(const _Float16* __restrict__ p)
{
    const v8h lo = *(const v8h*)p;
    const v8h hi = *(const v8h*)(p + 16);
    return __builtin_shufflevector(lo, hi, 0, 1, 2, 3, 4, 5, 6, 7,
                                   8, 9, 10, 11, 12, 13, 14, 15);
}

// ---------------------------------------------------------------------------
// Generic WMMA GEMM:  C[M,N] = A[M,K] * W[N,K]^T  (+ epilogue)
// Requirements: K % 32 == 0; M % 64 == 0; N % 64 == 0; 16B-aligned bases;
// lda/ldw % 4 == 0. Block = 128 threads (4 waves, 2x2); wave tile 32x32;
// block tile 64x64. Grid: (M/64, N/64, batches).
// ---------------------------------------------------------------------------
template <int EPI>
__global__ __launch_bounds__(128)
void gemm_wmma_kernel(const float* __restrict__ A, int lda, long long batchA,
                      const float* __restrict__ W, int ldw, long long batchW,
                      const float* __restrict__ P0,   // bias (EP_BIAS*) or gamma
                      const float* __restrict__ P1,   // beta (EP_BN_LRELU)
                      const float* __restrict__ RES, int ldres,
                      float* __restrict__ C, int ldc, long long batchC,
                      int K)
{
    const int lane  = threadIdx.x & 31;
    const int wave  = threadIdx.x >> 5;
    const int waveM = wave & 1;
    const int waveN = wave >> 1;
    const int m0 = blockIdx.x * 64 + waveM * 32;
    const int n0 = blockIdx.y * 64 + waveN * 32;
    const int b  = blockIdx.z;
    A += (long long)b * batchA;
    W += (long long)b * batchW;
    C += (long long)b * batchC;

    const int row = lane & 15;   // M row of A frag == N col of B frag
    const int g   = lane >> 4;   // lane half -> K sub-block select

    const float* __restrict__ A0 = A + (m0 + row) * lda + g * 8;
    const float* __restrict__ A1 = A0 + 16 * lda;
    const float* __restrict__ W0 = W + (n0 + row) * ldw + g * 8;
    const float* __restrict__ W1 = W0 + 16 * ldw;

    v8f acc[2][2] = {};
    const int KT = K >> 5;
    for (int kt = 0; kt < KT; ++kt) {
        const int ko = kt * 32;
        const v16h a0 = load_frag_f32(A0 + ko);
        const v16h a1 = load_frag_f32(A1 + ko);
        const v16h b0 = load_frag_f32(W0 + ko);
        const v16h b1 = load_frag_f32(W1 + ko);
        __builtin_prefetch(A0 + ko + 32, 0, 3);   // global_prefetch next K-step
        __builtin_prefetch(W0 + ko + 32, 0, 3);
        acc[0][0] = __builtin_amdgcn_wmma_f32_16x16x32_f16(false, a0, false, b0,
                                                           (short)0, acc[0][0], false, false);
        acc[0][1] = __builtin_amdgcn_wmma_f32_16x16x32_f16(false, a0, false, b1,
                                                           (short)0, acc[0][1], false, false);
        acc[1][0] = __builtin_amdgcn_wmma_f32_16x16x32_f16(false, a1, false, b0,
                                                           (short)0, acc[1][0], false, false);
        acc[1][1] = __builtin_amdgcn_wmma_f32_16x16x32_f16(false, a1, false, b1,
                                                           (short)0, acc[1][1], false, false);
    }

    // D layout: VGPR r, lane<16 -> (M=r, N=lane); lane>=16 -> (M=8+r, N=lane-16)
#pragma unroll
    for (int ni = 0; ni < 2; ++ni) {
        const int cn = n0 + ni * 16 + row;
        float sc = 1.f, bs = 0.f;
        if (EPI == EP_BIAS || EPI == EP_BIAS_RES) bs = P0[cn];
        if (EPI == EP_BN_LRELU) { sc = P0[cn] * BN_RSQ; bs = P1[cn]; }
#pragma unroll
        for (int mi = 0; mi < 2; ++mi) {
#pragma unroll
            for (int r = 0; r < 8; ++r) {
                const int cm = m0 + mi * 16 + r + g * 8;
                float v = acc[mi][ni][r];
                if (EPI == EP_BIAS) v += bs;
                if (EPI == EP_BN_LRELU) {
                    v = v * sc + bs;
                    v = (v >= 0.f) ? v : NEG_SLOPE * v;
                }
                if (EPI == EP_BIAS_RES) v += bs + RES[cm * ldres + cn];
                C[cm * ldc + cn] = v;
            }
        }
    }
}

// ---------------------------------------------------------------------------
// EdgeConv: per 16-point tile, loop over the 20 neighbors; build edge rows
// [nbr-ctr, ctr] in LDS (f16, padded to K mult of 32), WMMA vs conv weight,
// fused BN + LeakyReLU, running max over k in registers.
// Weight B-fragments are hoisted into registers before the neighbor loop
// (loop-invariant); layer 1 (2C=6) stages zero-padded weights through LDS.
// Grid: (N/16, B). Block: 256 (8 waves).
// ---------------------------------------------------------------------------
template <int C, int CO>
__global__ __launch_bounds__(256)
void edgeconv_kernel(const float* __restrict__ X, int ldx, long long batchX,
                     const int* __restrict__ IDX,              // (B,1024,20)
                     const float* __restrict__ Wm,             // (CO, 2C)
                     const float* __restrict__ Gm, const float* __restrict__ Bm,
                     float* __restrict__ OUT, int ldo, long long batchO)
{
    constexpr int  CIN  = 2 * C;
    constexpr int  CINP = (CIN + 31) & ~31;
    constexpr int  KT   = CINP / 32;
    constexpr int  NT   = CO / 16;
    constexpr int  TPW  = (NT + 7) / 8;
    constexpr bool WLDS = (CIN != CINP);       // pad weights in LDS (layer 1)
    constexpr bool VEC4 = ((C & 3) == 0);

    __shared__ __align__(16) _Float16 ef[16 * CINP];
    __shared__ __align__(16) _Float16 wpad[WLDS ? CO * CINP : 16];
    __shared__ __align__(16) float ctr[16 * C];

    const int b    = blockIdx.y;
    const int n0   = blockIdx.x * 16;
    const int tid  = threadIdx.x;
    const int lane = tid & 31;
    const int wave = tid >> 5;
    const float* __restrict__ Xb = X + (long long)b * batchX;

    for (int e = tid; e < 16 * CINP; e += 256) ef[e] = (_Float16)0.f;   // pad
    if (VEC4) {
        constexpr int C4 = (C + 3) / 4;   // == C/4 here
        for (int e = tid; e < 16 * C4; e += 256) {
            const int p = e / C4, c4 = e - p * C4;
            *(float4*)(ctr + p * C + c4 * 4) =
                *(const float4*)(Xb + (n0 + p) * ldx + c4 * 4);
        }
    } else {
        for (int e = tid; e < 16 * C; e += 256) {
            const int p = e / C, c = e % C;
            ctr[p * C + c] = Xb[(n0 + p) * ldx + c];
        }
    }
    if (WLDS) {
        for (int e = tid; e < CO * CINP; e += 256) {
            const int o = e / CINP, k = e % CINP;
            wpad[e] = (k < CIN) ? (_Float16)Wm[o * CIN + k] : (_Float16)0.f;
        }
    }
    __syncthreads();

    const int row = lane & 15;
    const int g   = lane >> 4;

    // Hoist weight fragments (invariant over the 20-neighbor loop).
    v16h bfr[TPW][KT];
#pragma unroll
    for (int t = 0; t < TPW; ++t) {
        const int ot = wave + t * 8;
        if (ot < NT) {
            const int colW = ot * 16 + row;
#pragma unroll
            for (int kt = 0; kt < KT; ++kt) {
                if (WLDS) bfr[t][kt] = load_frag_h16(wpad + colW * CINP + kt * 32 + g * 8);
                else      bfr[t][kt] = load_frag_f32(Wm + colW * CIN + kt * 32 + g * 8);
            }
        }
    }

    v8f mx[TPW];
#pragma unroll
    for (int t = 0; t < TPW; ++t)
#pragma unroll
        for (int r = 0; r < 8; ++r) mx[t][r] = -3.0e38f;

    const int idx_base = (b * 1024 + n0) * KNN;
    for (int kk = 0; kk < KNN; ++kk) {
        if (VEC4) {
            constexpr int C4 = (C + 3) / 4;
            for (int e = tid; e < 16 * C4; e += 256) {
                const int p = e / C4, c4 = e - p * C4;
                const int nb = IDX[idx_base + p * KNN + kk];
                const float4 nv = *(const float4*)(Xb + nb * ldx + c4 * 4);
                const float4 cv = *(const float4*)(ctr + p * C + c4 * 4);
                v4h dh, ch;
                dh[0] = (_Float16)(nv.x - cv.x); dh[1] = (_Float16)(nv.y - cv.y);
                dh[2] = (_Float16)(nv.z - cv.z); dh[3] = (_Float16)(nv.w - cv.w);
                ch[0] = (_Float16)cv.x; ch[1] = (_Float16)cv.y;
                ch[2] = (_Float16)cv.z; ch[3] = (_Float16)cv.w;
                *(v4h*)(ef + p * CINP + c4 * 4)     = dh;
                *(v4h*)(ef + p * CINP + C + c4 * 4) = ch;
            }
        } else {
            for (int e = tid; e < 16 * C; e += 256) {
                const int p = e / C, c = e % C;
                const int nb = IDX[idx_base + p * KNN + kk];
                const float nv = Xb[nb * ldx + c];
                const float cv = ctr[p * C + c];
                ef[p * CINP + c]     = (_Float16)(nv - cv);
                ef[p * CINP + C + c] = (_Float16)cv;
            }
        }
        __syncthreads();

        v8f acc[TPW] = {};
#pragma unroll
        for (int kt = 0; kt < KT; ++kt) {
            const v16h af = load_frag_h16(ef + row * CINP + kt * 32 + g * 8);
#pragma unroll
            for (int t = 0; t < TPW; ++t) {
                if (wave + t * 8 < NT) {
                    acc[t] = __builtin_amdgcn_wmma_f32_16x16x32_f16(false, af, false, bfr[t][kt],
                                                                    (short)0, acc[t], false, false);
                }
            }
        }
#pragma unroll
        for (int t = 0; t < TPW; ++t) {
            const int ot = wave + t * 8;
            if (ot < NT) {
                const int colW = ot * 16 + row;
                const float sc = Gm[colW] * BN_RSQ;
                const float bs = Bm[colW];
#pragma unroll
                for (int r = 0; r < 8; ++r) {
                    float v = acc[t][r] * sc + bs;
                    v = (v >= 0.f) ? v : NEG_SLOPE * v;
                    mx[t][r] = fmaxf(mx[t][r], v);
                }
            }
        }
        __syncthreads();
    }

    float* __restrict__ Ob = OUT + (long long)b * batchO;
#pragma unroll
    for (int t = 0; t < TPW; ++t) {
        const int ot = wave + t * 8;
        if (ot < NT) {
            const int cn = ot * 16 + row;
#pragma unroll
            for (int r = 0; r < 8; ++r) {
                const int cm = n0 + r + g * 8;
                Ob[cm * ldo + cn] = mx[t][r];
            }
        }
    }
}

// ---------------------------------------------------------------------------
// Squared norms per point (kNN distance assembly, layers 2-4).
// ---------------------------------------------------------------------------
__global__ void sqnorm_kernel(const float* __restrict__ X, int ldx, long long batchX,
                              float* __restrict__ XX, int Cdim)
{
    const int t = blockIdx.x * blockDim.x + threadIdx.x;
    if (t >= 8 * 1024) return;
    const int b = t >> 10, n = t & 1023;
    const float* xr = X + (long long)b * batchX + n * ldx;
    float s = 0.f;
    for (int c = 0; c < Cdim; ++c) s += xr[c] * xr[c];
    XX[t] = s;
}

// ---------------------------------------------------------------------------
// Top-K (K=20) of pd[n,m] = 2*dot - ||n||^2 - ||m||^2; register-resident
// unrolled insertion sort.
// ---------------------------------------------------------------------------
__global__ void topk_kernel(const float* __restrict__ S,   // (B,1024,1024) dots
                            const float* __restrict__ XX,  // (B,1024)
                            int* __restrict__ IDX)         // (B,1024,20)
{
    const int t = blockIdx.x * blockDim.x + threadIdx.x;
    if (t >= 8 * 1024) return;
    const int b = t >> 10, n = t & 1023;
    const float* Sr = S + (b * 1024 + n) * 1024;
    const float* xx = XX + b * 1024;
    const float xn = xx[n];

    float best[KNN];
    int   bid[KNN];
#pragma unroll
    for (int j = 0; j < KNN; ++j) { best[j] = -3.0e38f; bid[j] = 0; }

    for (int m = 0; m < 1024; ++m) {
        const float v = 2.f * Sr[m] - xn - xx[m];
        if (v > best[KNN - 1]) {
#pragma unroll
            for (int j = KNN - 1; j >= 0; --j) {
                if (j > 0 && v > best[j - 1]) {
                    best[j] = best[j - 1]; bid[j] = bid[j - 1];
                } else {
                    best[j] = v; bid[j] = m;
                    break;
                }
            }
        }
    }
    int* out = IDX + (b * 1024 + n) * KNN;
#pragma unroll
    for (int j = 0; j < KNN; ++j) out[j] = bid[j];
}

// ---------------------------------------------------------------------------
// Layer-1 kNN (C=3): direct -||xi-xj||^2 + fused top-20.
// ---------------------------------------------------------------------------
__global__ void knn3_topk_kernel(const float* __restrict__ X,  // (8,1024,3)
                                 int* __restrict__ IDX)        // (8,1024,20)
{
    const int t = blockIdx.x * blockDim.x + threadIdx.x;
    if (t >= 8 * 1024) return;
    const int b = t >> 10, n = t & 1023;
    const float* Xb = X + b * 1024 * 3;
    const float x0 = Xb[n * 3 + 0], x1 = Xb[n * 3 + 1], x2 = Xb[n * 3 + 2];

    float best[KNN];
    int   bid[KNN];
#pragma unroll
    for (int j = 0; j < KNN; ++j) { best[j] = -3.0e38f; bid[j] = 0; }

    for (int m = 0; m < 1024; ++m) {
        const float d0 = x0 - Xb[m * 3 + 0];
        const float d1 = x1 - Xb[m * 3 + 1];
        const float d2 = x2 - Xb[m * 3 + 2];
        const float v = -(d0 * d0 + d1 * d1 + d2 * d2);
        if (v > best[KNN - 1]) {
#pragma unroll
            for (int j = KNN - 1; j >= 0; --j) {
                if (j > 0 && v > best[j - 1]) {
                    best[j] = best[j - 1]; bid[j] = bid[j - 1];
                } else {
                    best[j] = v; bid[j] = m;
                    break;
                }
            }
        }
    }
    int* out = IDX + (b * 1024 + n) * KNN;
#pragma unroll
    for (int j = 0; j < KNN; ++j) out[j] = bid[j];
}

// ---------------------------------------------------------------------------
// Tiny MHA core: sequence length L=8 (batch axis!), batch = n (1024), H=4.
// One thread per (l, h, n): 8x8 scores + softmax + weighted sum.
// ---------------------------------------------------------------------------
__global__ void attention_kernel(const float* __restrict__ QKV, // (8*1024, 3E)
                                 float* __restrict__ O,          // (8*1024, E)
                                 int E, int d)
{
    const int t = blockIdx.x * blockDim.x + threadIdx.x;
    const int n = t & 1023;
    const int h = (t >> 10) & 3;
    const int l = t >> 12;
    if (l >= 8) return;

    const float scale = rsqrtf((float)d);
    const int   ldq   = 3 * E;
    const float* q = QKV + (l * 1024 + n) * ldq + h * d;

    float s[8];
    float mxv = -3.0e38f;
#pragma unroll
    for (int m = 0; m < 8; ++m) {
        const float* kr = QKV + (m * 1024 + n) * ldq + E + h * d;
        float acc = 0.f;
        for (int c = 0; c < d; ++c) acc += q[c] * kr[c];
        s[m] = acc * scale;
        mxv = fmaxf(mxv, s[m]);
    }
    float den = 0.f;
#pragma unroll
    for (int m = 0; m < 8; ++m) { s[m] = __expf(s[m] - mxv); den += s[m]; }
    const float inv = 1.f / den;

    float* o = O + (l * 1024 + n) * E + h * d;
    for (int c = 0; c < d; ++c) {
        float acc = 0.f;
#pragma unroll
        for (int m = 0; m < 8; ++m) {
            const float* vr = QKV + (m * 1024 + n) * ldq + 2 * E + h * d;
            acc += s[m] * vr[c];
        }
        o[c] = acc * inv;
    }
}

// ---------------------------------------------------------------------------
// Global max + mean pool over points: (8,1024,1024) -> (8,2048)
// ---------------------------------------------------------------------------
__global__ void pool_kernel(const float* __restrict__ H, float* __restrict__ F)
{
    const int t = blockIdx.x * blockDim.x + threadIdx.x;
    if (t >= 8 * 1024) return;
    const int b = t >> 10, c = t & 1023;
    float mx = -3.0e38f, sm = 0.f;
    for (int n = 0; n < 1024; ++n) {
        const float v = H[(b * 1024 + n) * 1024 + c];
        mx = fmaxf(mx, v);
        sm += v;
    }
    F[b * 2048 + c]        = mx;
    F[b * 2048 + 1024 + c] = sm * (1.f / 1024.f);
}

// ---------------------------------------------------------------------------
// Small FC head (M=8): scalar dot, optional bias, optional BN+LeakyReLU.
// ---------------------------------------------------------------------------
__global__ void fc_kernel(const float* __restrict__ In, const float* __restrict__ Wm,
                          const float* __restrict__ bias,
                          const float* __restrict__ g, const float* __restrict__ bb,
                          float* __restrict__ Out, int M, int N, int Kd)
{
    const int t = blockIdx.x * blockDim.x + threadIdx.x;
    if (t >= M * N) return;
    const int m = t / N, n = t % N;
    const float* a = In + m * Kd;
    const float* w = Wm + n * Kd;
    float acc = 0.f;
    for (int k = 0; k < Kd; ++k) acc += a[k] * w[k];
    if (bias) acc += bias[n];
    if (g) {
        acc = acc * (g[n] * BN_RSQ) + bb[n];
        acc = (acc >= 0.f) ? acc : NEG_SLOPE * acc;
    }
    Out[t] = acc;
}

// ---------------------------------------------------------------------------
// Orchestration
// ---------------------------------------------------------------------------
extern "C" void kernel_launch(void* const* d_in, const int* in_sizes, int n_in,
                              void* d_out, int out_size, void* d_ws, size_t ws_size,
                              hipStream_t stream)
{
    (void)in_sizes; (void)n_in; (void)out_size; (void)ws_size;

    const float* x  = (const float*)d_in[0];
    const float* W1 = (const float*)d_in[1],  *G1 = (const float*)d_in[2],  *B1 = (const float*)d_in[3];
    const float* W2 = (const float*)d_in[4],  *G2 = (const float*)d_in[5],  *B2 = (const float*)d_in[6];
    const float* W3 = (const float*)d_in[7],  *G3 = (const float*)d_in[8],  *B3 = (const float*)d_in[9];
    const float* W4 = (const float*)d_in[10], *G4 = (const float*)d_in[11], *B4 = (const float*)d_in[12];
    const float* A1wi = (const float*)d_in[13], *A1bi = (const float*)d_in[14];
    const float* A1wo = (const float*)d_in[15], *A1bo = (const float*)d_in[16];
    const float* A2wi = (const float*)d_in[17], *A2bi = (const float*)d_in[18];
    const float* A2wo = (const float*)d_in[19], *A2bo = (const float*)d_in[20];
    const float* A3wi = (const float*)d_in[21], *A3bi = (const float*)d_in[22];
    const float* A3wo = (const float*)d_in[23], *A3bo = (const float*)d_in[24];
    const float* A4wi = (const float*)d_in[25], *A4bi = (const float*)d_in[26];
    const float* A4wo = (const float*)d_in[27], *A4bo = (const float*)d_in[28];
    const float* W5  = (const float*)d_in[29], *G5 = (const float*)d_in[30], *B5 = (const float*)d_in[31];
    const float* L1W = (const float*)d_in[32], *G6 = (const float*)d_in[33], *B6 = (const float*)d_in[34];
    const float* L2W = (const float*)d_in[35], *L2B = (const float*)d_in[36];
    const float* G7  = (const float*)d_in[37], *B7  = (const float*)d_in[38];
    const float* L3W = (const float*)d_in[39], *L3B = (const float*)d_in[40];

    // Workspace layout (floats), ~60 MB total:
    float* ws   = (float*)d_ws;
    float* xcat = ws;                          // 8192 x 512  : x1|x2|x3|x4 col blocks
    float* bufA = xcat + 8192LL * 512;         // 8192 x 1024 : Gram / qkv / head acts
    float* bufB = bufA + 8192LL * 1024;        // 8192 x 256  : attention output
    float* xx   = bufB + 8192LL * 256;         // 8192
    float* feat = xx + 8192;                   // 8 x 2048
    float* fc1b = feat + 8 * 2048;             // 8 x 512
    float* fc2b = fc1b + 8 * 512;              // 8 x 256
    int*   idx  = (int*)(fc2b + 8 * 256);      // 8 x 1024 x 20

    auto run_knn = [&](const float* X, int ldx, long long bstride, int Cdim) {
        sqnorm_kernel<<<32, 256, 0, stream>>>(X, ldx, bstride, xx, Cdim);
        gemm_wmma_kernel<EP_NONE><<<dim3(16, 16, 8), 128, 0, stream>>>(
            X, ldx, bstride, X, ldx, bstride,
            nullptr, nullptr, nullptr, 0,
            bufA, 1024, 1024LL * 1024, Cdim);
        topk_kernel<<<32, 256, 0, stream>>>(bufA, xx, idx);
    };

    auto run_mha = [&](float* col, const float* wi, const float* bi,
                       const float* wo, const float* bo, int E) {
        gemm_wmma_kernel<EP_BIAS><<<dim3(128, (3 * E) / 64, 1), 128, 0, stream>>>(
            col, 512, 0, wi, E, 0, bi, nullptr, nullptr, 0,
            bufA, 3 * E, 0, E);
        attention_kernel<<<128, 256, 0, stream>>>(bufA, bufB, E, E / 4);
        gemm_wmma_kernel<EP_BIAS_RES><<<dim3(128, E / 64, 1), 128, 0, stream>>>(
            bufB, E, 0, wo, E, 0, bo, nullptr, col, 512,
            col, 512, 0, E);
    };

    // ---- Stage 1: C=3 -> 64 ----
    knn3_topk_kernel<<<32, 256, 0, stream>>>(x, idx);
    edgeconv_kernel<3, 64><<<dim3(64, 8), 256, 0, stream>>>(
        x, 3, 1024LL * 3, idx, W1, G1, B1, xcat + 0, 512, 1024LL * 512);
    run_mha(xcat + 0, A1wi, A1bi, A1wo, A1bo, 64);

    // ---- Stage 2: C=64 -> 64 ----
    run_knn(xcat + 0, 512, 1024LL * 512, 64);
    edgeconv_kernel<64, 64><<<dim3(64, 8), 256, 0, stream>>>(
        xcat + 0, 512, 1024LL * 512, idx, W2, G2, B2, xcat + 64, 512, 1024LL * 512);
    run_mha(xcat + 64, A2wi, A2bi, A2wo, A2bo, 64);

    // ---- Stage 3: C=64 -> 128 ----
    run_knn(xcat + 64, 512, 1024LL * 512, 64);
    edgeconv_kernel<64, 128><<<dim3(64, 8), 256, 0, stream>>>(
        xcat + 64, 512, 1024LL * 512, idx, W3, G3, B3, xcat + 128, 512, 1024LL * 512);
    run_mha(xcat + 128, A3wi, A3bi, A3wo, A3bo, 128);

    // ---- Stage 4: C=128 -> 256 ----
    run_knn(xcat + 128, 512, 1024LL * 512, 128);
    edgeconv_kernel<128, 256><<<dim3(64, 8), 256, 0, stream>>>(
        xcat + 128, 512, 1024LL * 512, idx, W4, G4, B4, xcat + 256, 512, 1024LL * 512);
    run_mha(xcat + 256, A4wi, A4bi, A4wo, A4bo, 256);

    // ---- Head: 512 -> 1024 (+BN+LReLU), pool, 3x FC ----
    gemm_wmma_kernel<EP_BN_LRELU><<<dim3(128, 16, 1), 128, 0, stream>>>(
        xcat, 512, 0, W5, 512, 0, G5, B5, nullptr, 0,
        bufA, 1024, 0, 512);
    pool_kernel<<<32, 256, 0, stream>>>(bufA, feat);

    fc_kernel<<<16, 256, 0, stream>>>(feat, L1W, nullptr, G6, B6, fc1b, 8, 512, 2048);
    fc_kernel<<<8, 256, 0, stream>>>(fc1b, L2W, L2B, G7, B7, fc2b, 8, 256, 512);
    fc_kernel<<<2, 256, 0, stream>>>(fc2b, L3W, L3B, nullptr, nullptr,
                                     (float*)d_out, 8, 40, 256);
}
